// GlobalLocalAttention_8632884265077
// MI455X (gfx1250) — matrix operand order repaired and checked by reading, not compiled, not run on hardware
//
#include <hip/hip_runtime.h>
#include <hip/hip_bf16.h>
#include <stdint.h>

typedef __attribute__((ext_vector_type(16))) _Float16 v16h;
typedef __attribute__((ext_vector_type(8)))  float    v8f;

#define C_   256
#define H_   128
#define W_   128
#define HW_  16384
#define B_   2
#define NH_  16
#define WS_  8
#define EPS_ 1e-5f

union AFrag  { v16h h; uint32_t u[8]; };
union Pack16 { _Float16 h[16]; uint4 v[2]; };

// ---------------------------------------------------------------------------
// Generic fused pointwise-conv GEMM:  Y[b, ch_off+o, n] = epilogue( sum_c W[o,c] * X[b, c*cstep, n] )
// epilogue: +bias, BN(scale/shift from stacked gamma/beta/mean/var), relu6, +gc[b, ch_off+o]
// Block: 256 thr = 8 waves; tile 64M x 64N; K-step 64 (4 WMMA / barrier pair);
// f16 WMMA, f32 accum. All LDS staging uses 128-bit conflict-free stores.
// ---------------------------------------------------------------------------
__global__ __launch_bounds__(256) void k_pw_gemm(
    const float* __restrict__ X, const float* __restrict__ Wt,
    const float* __restrict__ bias, const float* __restrict__ bn,
    const float* __restrict__ gc, float* __restrict__ Y,
    int Cin, long in_bstride, int in_cstride,
    int Co_tot, int ch_off, int relu6)
{
  const int N = HW_;
  const int b  = blockIdx.z;
  const int n0 = blockIdx.x * 64;
  const int m0 = blockIdx.y * 64;
  const int Co = gridDim.y * 64;
  const float* Xb = X + (long)b * in_bstride;

  __shared__ _Float16 As[64][72];   // [m][k], pitch 144B (16B-aligned, 9-bank groups)
  __shared__ _Float16 Bs[64][72];   // [n][k]

  const int tid  = threadIdx.x;
  const int lane = tid & 31;
  const int wave = tid >> 5;
  const int wm = (wave & 3) * 16;   // wave M offset in block tile
  const int wn = (wave >> 2) * 32;  // wave N offset (each wave does 2 N-subtiles)

  v8f acc0 = {0.f,0.f,0.f,0.f,0.f,0.f,0.f,0.f};
  v8f acc1 = {0.f,0.f,0.f,0.f,0.f,0.f,0.f,0.f};

  const int ra = tid >> 2, ca = (tid & 3) * 16;   // A staging: 64 m x 64 k, 16 k per thread
  const int nb = tid & 63, kb = (tid >> 6) * 16;  // B staging: 64 n x 64 k, 16 k per thread

  for (int k0 = 0; k0 < Cin; k0 += 64) {
    // stage A (weights, row-major [Co][Cin]) -> f16, two b128 LDS stores
    {
      const float4* wp = (const float4*)(Wt + (long)(m0 + ra) * Cin + k0 + ca);
      Pack16 ap;
      #pragma unroll
      for (int q = 0; q < 4; ++q) {
        float4 w = wp[q];
        ap.h[q*4+0] = (_Float16)w.x; ap.h[q*4+1] = (_Float16)w.y;
        ap.h[q*4+2] = (_Float16)w.z; ap.h[q*4+3] = (_Float16)w.w;
      }
      *(uint4*)&As[ra][ca]     = ap.v[0];
      *(uint4*)&As[ra][ca + 8] = ap.v[1];
    }
    // stage B (activations): gather 16 channels of one pixel (coalesced across
    // the 64 threads sharing each channel row), pack, two b128 LDS stores.
    {
      const float* xp = Xb + (long)(k0 + kb) * in_cstride + n0 + nb;
      Pack16 bp;
      #pragma unroll
      for (int j = 0; j < 16; ++j) bp.h[j] = (_Float16)xp[(long)j * in_cstride];
      *(uint4*)&Bs[nb][kb]     = bp.v[0];
      *(uint4*)&Bs[nb][kb + 8] = bp.v[1];
    }
    __syncthreads();

    const uint32_t* arow = (const uint32_t*)&As[wm +      (lane & 15)][0];
    const uint32_t* br0  = (const uint32_t*)&Bs[wn +      (lane & 15)][0];
    const uint32_t* br1  = (const uint32_t*)&Bs[wn + 16 + (lane & 15)][0];
    const int sa = (lane < 16) ? 0 : 8;  // A: lane<16 -> K 0..15, else 16..31
    const int sb = (lane < 16) ? 0 : 8;  // B: lane<16 -> K 0..15, else 16..31

    #pragma unroll
    for (int ks = 0; ks < 2; ++ks) {
      const int o16 = ks * 16;           // uint offset of the 32-half K sub-block
      AFrag a, bf0, bf1;
      #pragma unroll
      for (int i = 0; i < 8; ++i) { a.u[i] = arow[o16 + sa + i]; }
      #pragma unroll
      for (int i = 0; i < 8; ++i) { bf0.u[i] = br0[o16 + sb + i]; bf1.u[i] = br1[o16 + sb + i]; }
      acc0 = __builtin_amdgcn_wmma_f32_16x16x32_f16(false, a.h, false, bf0.h, (short)0, acc0, false, false);
      acc1 = __builtin_amdgcn_wmma_f32_16x16x32_f16(false, a.h, false, bf1.h, (short)0, acc1, false, false);
    }
    __syncthreads();
  }

  // epilogue: D layout — VGPR r, lane l: M = r + 8*(l/16), N = l%16
  #pragma unroll
  for (int r = 0; r < 8; ++r) {
    const int M = r + ((lane >> 4) << 3);
    const int o = m0 + wm + M;
    float sc = 1.f, sh = 0.f;
    if (bn) {
      float g = bn[o], be = bn[Co + o], mu = bn[2*Co + o], va = bn[3*Co + o];
      sc = g * rsqrtf(va + EPS_);
      sh = be - mu * sc;
    }
    const float bo = bias ? bias[o] : 0.f;
    const float ga = gc ? gc[b * 1024 + ch_off + o] : 0.f;
    const long base = ((long)b * Co_tot + ch_off + o) * (long)N;
    const int nc = lane & 15;
    float v0 = (acc0[r] + bo) * sc + sh;
    float v1 = (acc1[r] + bo) * sc + sh;
    if (relu6) { v0 = fminf(fmaxf(v0, 0.f), 6.f); v1 = fminf(fmaxf(v1, 0.f), 6.f); }
    Y[base + n0 + wn +      nc] = v0 + ga;
    Y[base + n0 + wn + 16 + nc] = v1 + ga;
  }
}

// ---------------------------------------------------------------------------
// Depthwise KxK conv (zero pad, dilation) + per-channel bias
// ---------------------------------------------------------------------------
__global__ __launch_bounds__(256) void k_dw(
    const float* __restrict__ x, const float* __restrict__ w,
    const float* __restrict__ bias, float* __restrict__ y,
    int K, int pad, int dil)
{
  const int c = blockIdx.y, b = blockIdx.z;
  const int n = blockIdx.x * 256 + threadIdx.x;
  const int py = n >> 7, px = n & 127;
  const float* xp = x + ((long)b * C_ + c) * HW_;
  const float* wp = w + c * K * K;
  float acc = bias[c];
  for (int ky = 0; ky < K; ++ky) {
    const int iy = py - pad + ky * dil;
    if (iy < 0 || iy >= H_) continue;
    for (int kx = 0; kx < K; ++kx) {
      const int ix = px - pad + kx * dil;
      if (ix < 0 || ix >= W_) continue;
      acc += wp[ky * K + kx] * xp[iy * W_ + ix];
    }
  }
  y[((long)b * C_ + c) * HW_ + n] = acc;
}

// ---------------------------------------------------------------------------
// Global average pool over HxW  -> out[b*C + c]
// ---------------------------------------------------------------------------
__global__ __launch_bounds__(256) void k_gpool(const float* __restrict__ x, float* __restrict__ out)
{
  const int c = blockIdx.x, b = blockIdx.y;
  const float* xp = x + ((long)b * C_ + c) * HW_;
  float s = 0.f;
  for (int i = threadIdx.x; i < HW_; i += 256) s += xp[i];
  __shared__ float red[256];
  red[threadIdx.x] = s;
  __syncthreads();
  for (int st = 128; st > 0; st >>= 1) {
    if (threadIdx.x < st) red[threadIdx.x] += red[threadIdx.x + st];
    __syncthreads();
  }
  if (threadIdx.x == 0) out[b * C_ + c] = red[0] * (1.f / HW_);
}

// ---------------------------------------------------------------------------
// Tiny FC: out[b,o] = relu6(BN(sum_c W[o,c]*in[b,c] + bias[o]))  (gf/gc chain)
// ---------------------------------------------------------------------------
__global__ __launch_bounds__(64) void k_fc(
    const float* __restrict__ in, const float* __restrict__ w,
    const float* __restrict__ bias, const float* __restrict__ bn,
    float* __restrict__ out, int Cin)
{
  const int o = blockIdx.x * 64 + threadIdx.x;
  const int b = blockIdx.y;
  const int Co = gridDim.x * 64;
  const float* ip = in + b * Cin;
  const float* wp = w + (long)o * Cin;
  float acc = bias[o];
  for (int c = 0; c < Cin; ++c) acc += wp[c] * ip[c];
  float g = bn[o], be = bn[Co + o], mu = bn[2 * Co + o], va = bn[3 * Co + o];
  float sc = g * rsqrtf(va + EPS_);
  float v = acc * sc + (be - mu * sc);
  out[b * Co + o] = fminf(fmaxf(v, 0.f), 6.f);
}

// ---------------------------------------------------------------------------
// 8x8 window attention, one block per (head, window, batch), 64 thr = 1 row/thr
// ---------------------------------------------------------------------------
__global__ __launch_bounds__(64) void k_attn(
    const float* __restrict__ qkv, const float* __restrict__ rpb, float* __restrict__ av)
{
  const int h = blockIdx.x;
  const int wy = blockIdx.y >> 4, wx = blockIdx.y & 15;
  const int b = blockIdx.z;
  __shared__ float qs[64][17], ks[64][17], vs[64][17], ds[64][65];
  const int t = threadIdx.x;
  const int ty = t >> 3, tx = t & 7;
  const int gy = wy * 8 + ty, gx = wx * 8 + tx;
  const long pix = (long)gy * W_ + gx;
  const float* base = qkv + (long)b * 3 * C_ * HW_;
  #pragma unroll
  for (int d = 0; d < 16; ++d) {
    qs[t][d] = base[((long)(          h * 16 + d)) * HW_ + pix];
    ks[t][d] = base[((long)(C_      + h * 16 + d)) * HW_ + pix];
    vs[t][d] = base[((long)(2 * C_  + h * 16 + d)) * HW_ + pix];
  }
  __syncthreads();
  float q[16];
  #pragma unroll
  for (int d = 0; d < 16; ++d) q[d] = qs[t][d];
  float m = -1e30f;
  for (int j = 0; j < 64; ++j) {
    float dot = 0.f;
    #pragma unroll
    for (int d = 0; d < 16; ++d) dot += q[d] * ks[j][d];
    const int jy = j >> 3, jx = j & 7;
    const float bi = rpb[((ty - jy + 7) * 15 + (tx - jx + 7)) * NH_ + h];
    dot = dot * 0.25f + bi;     // SCALE = HD^-0.5 = 0.25
    ds[t][j] = dot;
    m = fmaxf(m, dot);
  }
  float ssum = 0.f;
  for (int j = 0; j < 64; ++j) { float e = expf(ds[t][j] - m); ds[t][j] = e; ssum += e; }
  const float inv = 1.f / ssum;
  #pragma unroll
  for (int d = 0; d < 16; ++d) {
    float a = 0.f;
    for (int j = 0; j < 64; ++j) a += ds[t][j] * vs[j][d];
    av[((long)b * C_ + h * 16 + d) * HW_ + pix] = a * inv;
  }
}

// ---------------------------------------------------------------------------
// out = avg8_vertical(reflect-pad bottom 1, zero pad 3) + avg8_horizontal(...) + local
// ---------------------------------------------------------------------------
__global__ __launch_bounds__(256) void k_merge(
    const float* __restrict__ av, const float* __restrict__ loc, float* __restrict__ out)
{
  const int c = blockIdx.y, b = blockIdx.z;
  const int n = blockIdx.x * 256 + threadIdx.x;
  const int py = n >> 7, px = n & 127;
  const float* ap = av + ((long)b * C_ + c) * HW_;
  float sx = 0.f, sy = 0.f;
  #pragma unroll
  for (int k = 0; k < 8; ++k) {
    int r = py - 3 + k;
    if (r >= 0 && r <= 128) { int rr = (r == 128) ? 126 : r; sx += ap[rr * W_ + px]; }
    int q = px - 3 + k;
    if (q >= 0 && q <= 128) { int qq = (q == 128) ? 126 : q; sy += ap[py * W_ + qq]; }
  }
  const long idx = ((long)b * C_ + c) * HW_ + n;
  out[idx] = sx * 0.125f + sy * 0.125f + loc[idx];
}

// ---------------------------------------------------------------------------
// Final projection depthwise 8x8: input logically reflect-padded to 129x129,
// conv pad 3 (zero outside padded tensor), fused BN. Index-mapped, no materialized pad.
// ---------------------------------------------------------------------------
__global__ __launch_bounds__(256) void k_proj_dw(
    const float* __restrict__ s, const float* __restrict__ w,
    const float* __restrict__ bn, float* __restrict__ y)
{
  const int c = blockIdx.y, b = blockIdx.z;
  const int n = blockIdx.x * 256 + threadIdx.x;
  const int py = n >> 7, px = n & 127;
  const float* sp = s + ((long)b * C_ + c) * HW_;
  const float* wp = w + c * 64;
  float acc = 0.f;
  for (int ky = 0; ky < 8; ++ky) {
    int iy = py - 3 + ky;
    if (iy < 0 || iy > 128) continue;
    iy = (iy == 128) ? 126 : iy;
    for (int kx = 0; kx < 8; ++kx) {
      int ix = px - 3 + kx;
      if (ix < 0 || ix > 128) continue;
      ix = (ix == 128) ? 126 : ix;
      acc += wp[ky * 8 + kx] * sp[iy * W_ + ix];
    }
  }
  const float g = bn[c], be = bn[C_ + c], mu = bn[2 * C_ + c], va = bn[3 * C_ + c];
  const float sc = g * rsqrtf(va + EPS_);
  y[((long)b * C_ + c) * HW_ + n] = acc * sc + (be - mu * sc);
}

// ---------------------------------------------------------------------------
extern "C" void kernel_launch(void* const* d_in, const int* in_sizes, int n_in,
                              void* d_out, int out_size, void* d_ws, size_t ws_size,
                              hipStream_t stream)
{
  (void)in_sizes; (void)n_in; (void)out_size; (void)ws_size;
  const float* x       = (const float*)d_in[0];
  const float* b1_w    = (const float*)d_in[1];
  const float* b1_b    = (const float*)d_in[2];
  const float* b1_bn   = (const float*)d_in[3];
  const float* b2_dw   = (const float*)d_in[4];
  const float* b2_dwb  = (const float*)d_in[5];
  const float* b2_pw   = (const float*)d_in[6];
  const float* b2_pwb  = (const float*)d_in[7];
  const float* b2_bn   = (const float*)d_in[8];
  const float* b3_dw   = (const float*)d_in[9];
  const float* b3_dwb  = (const float*)d_in[10];
  const float* b3_pw   = (const float*)d_in[11];
  const float* b3_pwb  = (const float*)d_in[12];
  const float* b3_bn   = (const float*)d_in[13];
  const float* b4_dw   = (const float*)d_in[14];
  const float* b4_dwb  = (const float*)d_in[15];
  const float* b4_pw   = (const float*)d_in[16];
  const float* b4_pwb  = (const float*)d_in[17];
  const float* b4_bn   = (const float*)d_in[18];
  const float* gf_w    = (const float*)d_in[19];
  const float* gf_b    = (const float*)d_in[20];
  const float* gf_bn   = (const float*)d_in[21];
  const float* gc_w    = (const float*)d_in[22];
  const float* gc_b    = (const float*)d_in[23];
  const float* gc_bn   = (const float*)d_in[24];
  const float* bc_w    = (const float*)d_in[25];
  const float* bc_b    = (const float*)d_in[26];
  const float* bc_bn   = (const float*)d_in[27];
  const float* ba_w    = (const float*)d_in[28];
  const float* ba_b    = (const float*)d_in[29];
  const float* ba_bn   = (const float*)d_in[30];
  const float* qkv_w   = (const float*)d_in[31];
  const float* rpb     = (const float*)d_in[32];
  const float* proj_dw = (const float*)d_in[33];
  const float* proj_bn = (const float*)d_in[34];
  const float* proj_pw = (const float*)d_in[35];
  float* out = (float*)d_out;

  // workspace layout (floats). Needs ~336 MB.
  float* ws       = (float*)d_ws;
  float* branch4  = ws;                                     // B*1024*HW
  float* bc_out   = branch4 + (size_t)B_ * 1024 * HW_;      // B*1024*HW
  float* tmpA     = bc_out  + (size_t)B_ * 1024 * HW_;      // B*C*HW (dw scratch, later merged sum)
  float* localb   = tmpA    + (size_t)B_ * C_   * HW_;      // B*C*HW
  float* gf_pool  = localb  + (size_t)B_ * C_   * HW_;      // B*C
  float* gf_act   = gf_pool + (size_t)B_ * C_;              // B*C
  float* gc_act   = gf_act  + (size_t)B_ * C_;              // B*4C
  float* qkvb     = branch4;                                // alias (branch4 dead by then)
  float* avb      = bc_out;                                 // alias (bc_out dead by then)
  float* pdw      = bc_out + (size_t)B_ * C_ * HW_;         // alias, disjoint from avb

  const long xbs = (long)C_ * HW_;       // batch stride of (B,C,H,W)
  const long bbs = (long)1024 * HW_;     // batch stride of (B,4C,H,W)

  dim3 blk256(256), blk64(64);
  dim3 gemm256(HW_ / 64, 4, B_);         // Co = 256
  dim3 gemm768(HW_ / 64, 12, B_);        // Co = 768
  dim3 gemm1024(HW_ / 64, 16, B_);       // Co = 1024
  dim3 plane(HW_ / 256, C_, B_);

  // global context chain (tiny)
  k_gpool<<<dim3(C_, B_), blk256, 0, stream>>>(x, gf_pool);
  k_fc<<<dim3(4, B_),  blk64, 0, stream>>>(gf_pool, gf_w, gf_b, gf_bn, gf_act, C_);
  k_fc<<<dim3(16, B_), blk64, 0, stream>>>(gf_act,  gc_w, gc_b, gc_bn, gc_act, C_);

  // branch 1: 1x1 conv on x
  k_pw_gemm<<<gemm256, blk256, 0, stream>>>(x, b1_w, b1_b, b1_bn, gc_act, branch4,
                                            C_, xbs, HW_, 1024, 0, 1);
  // branches 2..4: depthwise + pointwise
  k_dw<<<plane, blk256, 0, stream>>>(x, b2_dw, b2_dwb, tmpA, 3, 1, 1);
  k_pw_gemm<<<gemm256, blk256, 0, stream>>>(tmpA, b2_pw, b2_pwb, b2_bn, gc_act, branch4,
                                            C_, xbs, HW_, 1024, 256, 1);
  k_dw<<<plane, blk256, 0, stream>>>(x, b3_dw, b3_dwb, tmpA, 4, 3, 2);
  k_pw_gemm<<<gemm256, blk256, 0, stream>>>(tmpA, b3_pw, b3_pwb, b3_bn, gc_act, branch4,
                                            C_, xbs, HW_, 1024, 512, 1);
  k_dw<<<plane, blk256, 0, stream>>>(x, b4_dw, b4_dwb, tmpA, 5, 4, 2);
  k_pw_gemm<<<gemm256, blk256, 0, stream>>>(tmpA, b4_pw, b4_pwb, b4_bn, gc_act, branch4,
                                            C_, xbs, HW_, 1024, 768, 1);

  // big 1024x1024 fuse conv (dominant GEMM)
  k_pw_gemm<<<gemm1024, blk256, 0, stream>>>(branch4, bc_w, bc_b, bc_bn, nullptr, bc_out,
                                             1024, bbs, HW_, 1024, 0, 1);

  // local path: pixel-shuffle ::2,::2 select == channels 4c of bc_out -> cstep=4
  k_pw_gemm<<<gemm256, blk256, 0, stream>>>(bc_out, ba_w, ba_b, ba_bn, nullptr, localb,
                                            C_, bbs, 4 * HW_, 256, 0, 1);

  // attention path (branch4/bc_out now dead -> reuse)
  k_pw_gemm<<<gemm768, blk256, 0, stream>>>(x, qkv_w, nullptr, nullptr, nullptr, qkvb,
                                            C_, xbs, HW_, 768, 0, 0);
  k_attn<<<dim3(NH_, 256, B_), blk64, 0, stream>>>(qkvb, rpb, avb);

  // directional average pools + local add
  k_merge<<<plane, blk256, 0, stream>>>(avb, localb, tmpA);

  // final projection: reflect-pad dw8x8 + BN, then 1x1 conv -> d_out
  k_proj_dw<<<plane, blk256, 0, stream>>>(tmpA, proj_dw, proj_bn, pdw);
  k_pw_gemm<<<gemm256, blk256, 0, stream>>>(pdw, proj_pw, nullptr, nullptr, nullptr, out,
                                            C_, xbs, HW_, 256, 0, 0);
}
